// SwinTransformer_85701777424476
// MI455X (gfx1250) — matrix operand order repaired and checked by reading, not compiled, not run on hardware
//
#include <hip/hip_runtime.h>

#define BATCH 8
#define HH 64
#define WW 64
#define DIM 512
#define ODIM 256
#define LTOT 16384            // 4*H*W tokens per batch image
#define TILE_M 64             // output tokens per workgroup
#define KC 32                 // K chunk staged per GEMM step
#define LDSA_STRIDE 260       // u32 stride per A row (512 f16 = 256 DW + 4 pad)
#define LDSB_CSTRIDE 20       // u32 stride per B column (16 kp DW + 4 pad, 16B-aligned)

#if __has_builtin(__builtin_amdgcn_sched_barrier)
#define SCHED_BARRIER() __builtin_amdgcn_sched_barrier(0)
#else
#define SCHED_BARRIER() asm volatile("" ::: "memory")
#endif

typedef __attribute__((ext_vector_type(16))) _Float16 v16h;
typedef __attribute__((ext_vector_type(2)))  __fp16   v2fp16;   // return type of cvt_pkrtz
typedef __attribute__((ext_vector_type(8)))  float    v8f;

union Frag {
    uint4 q[2];
    v16h  h;
};

__device__ __forceinline__ unsigned int pack_h2(float a, float b) {
#if __has_builtin(__builtin_amdgcn_cvt_pkrtz)
    const v2fp16 p = __builtin_amdgcn_cvt_pkrtz(a, b);   // one v_cvt_pkrtz_f16_f32
    return __builtin_bit_cast(unsigned int, p);
#else
    unsigned short ha = __builtin_bit_cast(unsigned short, (_Float16)a);
    unsigned short hb = __builtin_bit_cast(unsigned short, (_Float16)b);
    return (unsigned int)ha | ((unsigned int)hb << 16);
#endif
}

__global__ __launch_bounds__(256)
void swin_expand_ln_reduce_pos(const float* __restrict__ x,
                               const float* __restrict__ gamma,
                               const float* __restrict__ beta,
                               const float* __restrict__ wred,
                               const float* __restrict__ alphap,
                               float* __restrict__ out) {
    __shared__ unsigned int ldsA[TILE_M * LDSA_STRIDE];        // 64 x 512 f16 pair-packed = 66.6KB
    __shared__ unsigned int ldsB[2][ODIM * LDSB_CSTRIDE];      // double-buffered, frag-major, 2x20KB

    const int tid  = threadIdx.x;
    const int wid  = tid >> 5;
    const int lane = tid & 31;

    // ---- tile coordinates ----------------------------------------------
    const int t  = blockIdx.x;            // 2048 tiles
    const int b  = t >> 8;                // / (16384/64)
    const int l0 = (t & 255) << 6;        // token offset within image
    const int i  = l0 >> 7;               // output row (0..127), fixed per tile
    const int j0 = l0 & 127;              // first output col (0 or 64)

    // bilinear row sources (half-pixel, clamped) — same for all 64 tokens
    const int kr = i >> 1;
    int rA, rB; float wyA, wyB;
    if (i & 1) { rA = kr;               rB = min(kr + 1, HH - 1); wyA = 0.75f; wyB = 0.25f; }
    else       { rA = max(kr - 1, 0);   rB = kr;                  wyA = 0.25f; wyB = 0.75f; }

    // ---- Phase A: upsample + LayerNorm -> f16 A-tile in LDS -------------
    for (int t2 = 0; t2 < TILE_M / 8; ++t2) {
        const int m = wid + (t2 << 3);           // token 0..63 (wave-interleaved)
        const int j = j0 + m;
        const int u = j >> 1;
        int cA, cB; float wxA, wxB;
        if (j & 1) { cA = u;             cB = min(u + 1, WW - 1); wxA = 0.75f; wxB = 0.25f; }
        else       { cA = max(u - 1, 0); cB = u;  wxA = 0.25f; wxB = 0.75f; }

        const float* p00 = x + (((size_t)(b * 4096 + rA * WW + cA)) << 9);
        const float* p01 = x + (((size_t)(b * 4096 + rA * WW + cB)) << 9);
        const float* p10 = x + (((size_t)(b * 4096 + rB * WW + cA)) << 9);
        const float* p11 = x + (((size_t)(b * 4096 + rB * WW + cB)) << 9);

        float4 vbuf[4];
        float s = 0.f, s2 = 0.f;
#pragma unroll
        for (int it = 0; it < 4; ++it) {         // 4 x float4 per lane, 512B/wave coalesced
            const int chb = (it << 7) + (lane << 2);
            const float4 a00 = *(const float4*)(p00 + chb);
            const float4 a01 = *(const float4*)(p01 + chb);
            const float4 a10 = *(const float4*)(p10 + chb);
            const float4 a11 = *(const float4*)(p11 + chb);
            float4 v;
            v.x = wyA * (wxA * a00.x + wxB * a01.x) + wyB * (wxA * a10.x + wxB * a11.x);
            v.y = wyA * (wxA * a00.y + wxB * a01.y) + wyB * (wxA * a10.y + wxB * a11.y);
            v.z = wyA * (wxA * a00.z + wxB * a01.z) + wyB * (wxA * a10.z + wxB * a11.z);
            v.w = wyA * (wxA * a00.w + wxB * a01.w) + wyB * (wxA * a10.w + wxB * a11.w);
            vbuf[it] = v;
            s  += (v.x + v.y) + (v.z + v.w);
            s2 += (v.x * v.x + v.y * v.y) + (v.z * v.z + v.w * v.w);
        }
#pragma unroll
        for (int off = 16; off > 0; off >>= 1) { // wave32 reduction
            s  += __shfl_xor(s,  off, 32);
            s2 += __shfl_xor(s2, off, 32);
        }
        const float mean = s * (1.f / 512.f);
        const float var  = s2 * (1.f / 512.f) - mean * mean;
        const float rstd = rsqrtf(var + 1e-5f);
#pragma unroll
        for (int it = 0; it < 4; ++it) {
            const int chb = (it << 7) + (lane << 2);
            const float4 g  = *(const float4*)(gamma + chb);
            const float4 bt = *(const float4*)(beta + chb);
            const float4 v  = vbuf[it];
            const float x0 = (v.x - mean) * rstd * g.x + bt.x;
            const float x1 = (v.y - mean) * rstd * g.y + bt.y;
            const float x2 = (v.z - mean) * rstd * g.z + bt.z;
            const float x3 = (v.w - mean) * rstd * g.w + bt.w;
            uint2 pk;
            pk.x = pack_h2(x0, x1);
            pk.y = pack_h2(x2, x3);
            *(uint2*)&ldsA[m * LDSA_STRIDE + (chb >> 1)] = pk;   // ds_store_b64
        }
    }

    // ---- stage first B chunk (frag-major: ldsB[buf][n*20 + kp]) ---------
    {
        unsigned int* buf = ldsB[0];
#pragma unroll
        for (int iter = 0; iter < 4; ++iter) {
            const int kp = (iter << 2) | (tid >> 6);
            const int n4 = (tid & 63) << 2;
            const float4 r0 = *(const float4*)&wred[(size_t)(2 * kp)     * ODIM + n4];
            const float4 r1 = *(const float4*)&wred[(size_t)(2 * kp + 1) * ODIM + n4];
            buf[(n4 + 0) * LDSB_CSTRIDE + kp] = pack_h2(r0.x, r1.x);
            buf[(n4 + 1) * LDSB_CSTRIDE + kp] = pack_h2(r0.y, r1.y);
            buf[(n4 + 2) * LDSB_CSTRIDE + kp] = pack_h2(r0.z, r1.z);
            buf[(n4 + 3) * LDSB_CSTRIDE + kp] = pack_h2(r0.w, r1.w);
        }
    }
    __syncthreads();

    // ---- Phase B: GEMM via v_wmma_f32_16x16x32_f16 ----------------------
    // wave tiling: 4 (M) x 2 (N); each wave: 16 rows x 128 cols = 8 accum tiles
    const int m0    = (wid >> 1) << 4;   // 0/16/32/48
    const int n0    = (wid & 1)  << 7;   // 0/128
    const int lhalf = lane >> 4;
    const int l16   = lane & 15;
    const int arow  = (m0 + l16) * LDSA_STRIDE + (lhalf << 2);
    const int bcol0 = (n0 + l16) * LDSB_CSTRIDE + (lhalf << 2);

    v8f acc0 = {0,0,0,0,0,0,0,0};
    v8f acc[8] = {acc0, acc0, acc0, acc0, acc0, acc0, acc0, acc0};

    for (int kc = 0; kc < DIM; kc += KC) {
        const int  cur     = (kc >> 5) & 1;
        const bool hasNext = (kc + KC < DIM);

        // (1) issue next-chunk global loads early: overlap the whole math block
        float4 s0[4], s1[4];
        if (hasNext) {
#pragma unroll
            for (int iter = 0; iter < 4; ++iter) {
                const int kp = (iter << 2) | (tid >> 6);
                const int n4 = (tid & 63) << 2;
                s0[iter] = *(const float4*)&wred[(size_t)(kc + KC + 2 * kp)     * ODIM + n4];
                s1[iter] = *(const float4*)&wred[(size_t)(kc + KC + 2 * kp + 1) * ODIM + n4];
            }
        }

        // (2) batch ALL fragment loads, then pin schedule: loads before WMMAs.
        //     Overlapping bf live ranges force distinct registers -> partial dscnt waits.
        Frag af;
        af.q[0] = *(const uint4*)&ldsA[arow + (kc >> 1)];
        af.q[1] = *(const uint4*)&ldsA[arow + (kc >> 1) + 8];

        const unsigned int* buf = ldsB[cur];
        Frag bf[8];
#pragma unroll
        for (int nt = 0; nt < 8; ++nt) {
            const int col = bcol0 + nt * (16 * LDSB_CSTRIDE);
            bf[nt].q[0] = *(const uint4*)&buf[col];
            bf[nt].q[1] = *(const uint4*)&buf[col + 8];
        }
        SCHED_BARRIER();
#pragma unroll
        for (int nt = 0; nt < 8; ++nt) {
            acc[nt] = __builtin_amdgcn_wmma_f32_16x16x32_f16(
                false, af.h, false, bf[nt].h, (short)0, acc[nt], false, false);
        }
        SCHED_BARRIER();

        // (3) convert + store staged chunk after the WMMA burst
        if (hasNext) {
            unsigned int* nxt = ldsB[cur ^ 1];
#pragma unroll
            for (int iter = 0; iter < 4; ++iter) {
                const int kp = (iter << 2) | (tid >> 6);
                const int n4 = (tid & 63) << 2;
                nxt[(n4 + 0) * LDSB_CSTRIDE + kp] = pack_h2(s0[iter].x, s1[iter].x);
                nxt[(n4 + 1) * LDSB_CSTRIDE + kp] = pack_h2(s0[iter].y, s1[iter].y);
                nxt[(n4 + 2) * LDSB_CSTRIDE + kp] = pack_h2(s0[iter].z, s1[iter].z);
                nxt[(n4 + 3) * LDSB_CSTRIDE + kp] = pack_h2(s0[iter].w, s1[iter].w);
            }
        }
        __syncthreads();   // staged next-buffer visible; current buffer free to overwrite
    }

    // ---- Epilogue: + alpha * sinusoidal 2D pos, NT-store f32 ------------
    // args are bounded (|e| <= 129 rad) -> native v_sin/v_cos are fine.
    // sec is wave-uniform per N-tile; y-grid sections have a row-invariant angle.
    const float alpha = alphap[0];
    const float kScale = 0.14619587897f;  // ln(10000)/63
#pragma unroll
    for (int nt = 0; nt < 8; ++nt) {
        const int n   = n0 + (nt << 4) + l16;
        const int dm  = n & 63;
        const int sec = n >> 6;            // 0:sin(j) 1:cos(j) 2:sin(i) 3:cos(i)
        const float f = __expf(-(float)dm * kScale);
        if (sec >= 2) {
            const float e  = (float)(i + 1) * f;
            const float pe = (sec & 1) ? __cosf(e) : __sinf(e);
            const float ap = alpha * pe;
#pragma unroll
            for (int r = 0; r < 8; ++r) {
                const int mt = m0 + r + (lhalf << 3);
                const size_t idx = (((size_t)(b * LTOT + l0 + mt)) << 8) + n;
                __builtin_nontemporal_store(acc[nt][r] + ap, &out[idx]);
            }
        } else {
#pragma unroll
            for (int r = 0; r < 8; ++r) {
                const int mt = m0 + r + (lhalf << 3);       // token within tile
                const int jj = j0 + mt;
                const float e  = (float)(jj + 1) * f;
                const float pe = (sec & 1) ? __cosf(e) : __sinf(e);
                const size_t idx = (((size_t)(b * LTOT + l0 + mt)) << 8) + n;
                __builtin_nontemporal_store(acc[nt][r] + alpha * pe, &out[idx]);
            }
        }
    }
}

extern "C" void kernel_launch(void* const* d_in, const int* in_sizes, int n_in,
                              void* d_out, int out_size, void* d_ws, size_t ws_size,
                              hipStream_t stream) {
    (void)in_sizes; (void)n_in; (void)d_ws; (void)ws_size; (void)out_size;
    const float* x     = (const float*)d_in[0];
    const float* gamma = (const float*)d_in[1];
    const float* beta  = (const float*)d_in[2];
    const float* wred  = (const float*)d_in[3];
    const float* alpha = (const float*)d_in[4];
    float* out = (float*)d_out;

    const int tiles = (BATCH * LTOT) / TILE_M;   // 2048 workgroups
    swin_expand_ln_reduce_pos<<<tiles, 256, 0, stream>>>(x, gamma, beta, wred, alpha, out);
}
